// CondConvNeXtBlock_46497315947267
// MI455X (gfx1250) — compile-verified, matrix operand order checked by reading, not run on hardware
//
#include <hip/hip_runtime.h>
#include <math.h>

// CondConvNeXtBlock for MI455X (gfx1250), wave32 + WMMA bf16 path.
// Workspace (~261 MB):
//   [0,64K)      kr   bf16 (L x C)      scaled K, row major
//   [64K,128K)   v    bf16 (L x C)
//   [128K..)     WqP, WoP (512 KB each), W1P, W2P (1.5 MB each):
//                weights pre-swizzled to bf16 WMMA-fragment panels
//   [+4.2M)      y    bf16 (B*T x C)    conv+AdaLN output
//   [+68M)       hid  bf16 (B*T x I)    GELU(y@W1+b1)
// Residual lives in d_out between k1 and k4.

#define Bc 16
#define Cc 512
#define Tc 4096
#define Lc 64
#define Hc 8
#define Ic 1536
#define DHc 64
#define SCALE 0.35355339059327373f  // (C/H)^(-1/4) = 64^-0.25

typedef __bf16 bf16;
typedef __attribute__((ext_vector_type(16))) __bf16 v16bf;
typedef __attribute__((ext_vector_type(8)))  __bf16 v8bf;
typedef __attribute__((ext_vector_type(2)))  __bf16 v2bf;
typedef __attribute__((ext_vector_type(8)))  float  v8f;

#define WMMA_BF16(a, b, c) \
  __builtin_amdgcn_wmma_f32_16x16x32_bf16(false, (a), false, (b), (short)0, (c), false, false)

// ---- fragment-contiguous layout ---------------------------------------------
// ISA 16-bit WMMA operand: lane ln holds n = ln&15, K in {kb..kb+7, kb+16..kb+23},
// kb = (ln>>4)*8.  Tiles stored as 32 lanes x 16 contiguous bf16 (32 B/lane) so a
// fragment load is one aligned v16bf (2x ds_load_b128).
// Inverse map: lh=(k>>3)&1, e=(k&7)|((k>>4)<<3), ln=lh*16+(n&15).
// 8 consecutive k (k%8==0) -> 8 contiguous e (same lh) -> 16 B vector staging.
__device__ __forceinline__ int frag_slot(int k, int nm) {
  return ((((k >> 3) & 1) << 4) + nm) * 16 + ((k & 7) + (((k >> 4) & 1) << 3));
}
__device__ __forceinline__ v16bf frag_ld(const bf16* s, int f, int lane) {
  return *(const v16bf*)(s + f * 512 + lane * 16);
}

// ---------------------------------------------------------------------------
// kp_w: pre-swizzle a fp32 weight (K x N, row major) into bf16 fragment
// panels: out[((k>>5)*(N/16) + (n>>4))*512 + frag_slot(k&31, n&15)]
// ---------------------------------------------------------------------------
__global__ __launch_bounds__(256) void kp_w(const float* __restrict__ W,
                                            bf16* __restrict__ o, int K, int N) {
  const int idx = blockIdx.x * 256 + threadIdx.x;  // k*N + n (n fast, coalesced)
  const int k = idx / N, n = idx % N;
  o[((size_t)((k >> 5) * (N >> 4) + (n >> 4)) << 9) + frag_slot(k & 31, n & 15)] =
      (bf16)W[idx];
}

// ---------------------------------------------------------------------------
// k0: kv = aux @ Wkv + bkv  -> kr (L x C, scaled) and v (L x C), bf16
// ---------------------------------------------------------------------------
__global__ __launch_bounds__(256) void k0_kv(const float* __restrict__ aux,
                                             const float* __restrict__ Wkv,
                                             const float* __restrict__ bkv,
                                             bf16* __restrict__ kr,
                                             bf16* __restrict__ vm) {
  const int idx = blockIdx.x * 256 + threadIdx.x;
  const int l = idx >> 10;
  const int j = idx & 1023;
  float acc = bkv[j];
  for (int c = 0; c < Cc; ++c)
    acc = fmaf(aux[l * Cc + c], Wkv[(size_t)c * (2 * Cc) + j], acc);
  if (j < Cc) kr[(size_t)l * Cc + j] = (bf16)(acc * SCALE);
  else        vm[(size_t)l * Cc + (j - Cc)] = (bf16)acc;
}

// ---------------------------------------------------------------------------
// k1: fused cross-attention + Wo projection + residual -> d_out (B,C,T)
// One block = one batch b, 32 rows of T. 256 threads = 8 waves.
// ---------------------------------------------------------------------------
__global__ __launch_bounds__(256) void k1_attn(const float* __restrict__ x,
                                               const bf16* __restrict__ WqP,
                                               const float* __restrict__ bq,
                                               const bf16* __restrict__ WoP,
                                               const float* __restrict__ bo,
                                               const bf16* __restrict__ kr,
                                               const bf16* __restrict__ vm,
                                               float* __restrict__ res) {
  __shared__ alignas(32) bf16 sQ[Cc * 32];   // 32 KB: scalar q [c][m], then GEMM2 A-fragments
  __shared__ alignas(32) bf16 sP[16 * 512];  // 16 KB: 16 B-fragments (32k x 256n panel)
  __shared__ alignas(32) bf16 sX[2 * 512];   //  2 KB: 2 A-fragments  (32k x 32m panel)

  const int b    = blockIdx.y;
  const int t0   = blockIdx.x * 32;
  const int tid  = threadIdx.x;
  const int lane = tid & 31;
  const int wave = tid >> 5;
  const float* xb = x + (size_t)b * Cc * Tc;

  const int mt   = wave & 1;   // A-fragment index (M tile)
  const int ngrp = wave >> 1;  // group of 4 N tiles

  // ---------------- GEMM1: q = (x^T @ Wq + bq) * SCALE -> sQ[c][m] --------
  {
    v8f acc[8] = {};
    for (int kk = 0; kk < 16; ++kk) {
      __syncthreads();
      {  // stage x^T panel: float4 along T (4 m per load), scatter 4 slots
        const int m4 = (tid & 7) * 4, k = tid >> 3;
        const float4 xv = *(const float4*)&xb[(size_t)(kk * 32 + k) * Tc + t0 + m4];
        const int base = (m4 >> 4) * 512 + frag_slot(k, m4 & 15);
        sX[base]      = (bf16)xv.x;
        sX[base + 16] = (bf16)xv.y;   // nm+1 -> slot+16
        sX[base + 32] = (bf16)xv.z;
        sX[base + 48] = (bf16)xv.w;
      }
      if (kk + 1 < 16)  // prefetch next pre-swizzled panel (global_prefetch_b8)
        __builtin_prefetch(WqP + (((size_t)(kk + 1) * 32) << 9) + tid * 64, 0, 1);
      for (int nh = 0; nh < 2; ++nh) {
        if (nh) __syncthreads();
        const bf16* src = WqP + (((size_t)kk * 32 + nh * 16) << 9);  // 16 KB panel
        for (int i = tid; i < 512; i += 256)
          *(v16bf*)(sP + i * 16) = *(const v16bf*)(src + i * 16);
        __syncthreads();
        v16bf af = frag_ld(sX, mt, lane);
#pragma unroll
        for (int j = 0; j < 4; ++j) {
          v16bf bfr = frag_ld(sP, ngrp * 4 + j, lane);
          acc[nh * 4 + j] = WMMA_BF16(af, bfr, acc[nh * 4 + j]);
        }
      }
    }
    // scatter accumulators (C-layout: m = mt*16 + 8*(lane>>4) + r, n = lane&15)
    const int mb = mt * 16 + 8 * (lane >> 4);
#pragma unroll
    for (int nh = 0; nh < 2; ++nh)
#pragma unroll
      for (int j = 0; j < 4; ++j) {
        const int c = nh * 256 + (ngrp * 4 + j) * 16 + (lane & 15);
        const float bqc = bq[c];
#pragma unroll
        for (int r = 0; r < 8; ++r)
          sQ[c * 32 + mb + r] = (bf16)((acc[nh * 4 + j][r] + bqc) * SCALE);
      }
  }
  __syncthreads();

  // -------- attention: thread (m = tid&31, h = tid>>5), flash softmax -----
  {
    const int m = tid & 31;
    const int h = tid >> 5;              // lane-uniform per wave
    const bf16* kh = kr + h * DHc;       // k[j*C + h*64 + d] (scaled)
    const bf16* vh = vm + h * DHc;       // v[j*C + h*64 + d]
    float qv[DHc], o[DHc];
#pragma unroll
    for (int d = 0; d < DHc; ++d) {
      qv[d] = (float)sQ[(h * DHc + d) * 32 + m];
      o[d]  = 0.f;
    }
    __syncthreads();  // all q reads done: sQ may now be overwritten (fragments)
    float mx = -1e30f, s = 0.f;
    for (int j = 0; j < Lc; ++j) {
      float ej = 0.f;
#pragma unroll
      for (int dv = 0; dv < DHc / 8; ++dv) {  // 16 B vector loads (lane-uniform)
        const v8bf kj = *(const v8bf*)&kh[(size_t)j * Cc + dv * 8];
#pragma unroll
        for (int e = 0; e < 8; ++e) ej = fmaf(qv[dv * 8 + e], (float)kj[e], ej);
      }
      const float nm = fmaxf(mx, ej);
      const float al = __expf(mx - nm);
      const float p  = __expf(ej - nm);
      s = s * al + p;
#pragma unroll
      for (int dv = 0; dv < DHc / 8; ++dv) {
        const v8bf vj = *(const v8bf*)&vh[(size_t)j * Cc + dv * 8];
#pragma unroll
        for (int e = 0; e < 8; ++e) {
          const int d = dv * 8 + e;
          o[d] = fmaf(p, (float)vj[e], o[d] * al);
        }
      }
      mx = nm;
    }
    const float inv = 1.f / s;
    // write attn-out directly as GEMM2 A-fragments (16 K-chunks x 2 M-tiles)
#pragma unroll
    for (int d = 0; d < DHc; ++d) {
      const int c = h * DHc + d;
      sQ[(c >> 5) * 1024 + (m >> 4) * 512 + frag_slot(c & 31, m & 15)] =
          (bf16)(o[d] * inv);
    }
  }
  __syncthreads();

  // ---------------- GEMM2: res = x + out @ Wo + bo ------------------------
  {
    v8f acc[8] = {};
    for (int kk = 0; kk < 16; ++kk) {
      v16bf af = frag_ld(sQ + kk * 1024, mt, lane);  // sQ stable during GEMM2
      if (kk + 1 < 16)
        __builtin_prefetch(WoP + (((size_t)(kk + 1) * 32) << 9) + tid * 64, 0, 1);
      for (int nh = 0; nh < 2; ++nh) {
        __syncthreads();
        const bf16* src = WoP + (((size_t)kk * 32 + nh * 16) << 9);
        for (int i = tid; i < 512; i += 256)
          *(v16bf*)(sP + i * 16) = *(const v16bf*)(src + i * 16);
        __syncthreads();
#pragma unroll
        for (int j = 0; j < 4; ++j) {
          v16bf bfr = frag_ld(sP, ngrp * 4 + j, lane);
          acc[nh * 4 + j] = WMMA_BF16(af, bfr, acc[nh * 4 + j]);
        }
      }
    }
    const int mb = mt * 16 + 8 * (lane >> 4);
#pragma unroll
    for (int nh = 0; nh < 2; ++nh)
#pragma unroll
      for (int j = 0; j < 4; ++j) {
        const int c = nh * 256 + (ngrp * 4 + j) * 16 + (lane & 15);
        const float boc = bo[c];
        const size_t base = (size_t)c * Tc + t0 + mb;  // 32 B aligned
        const float4 x0 = *(const float4*)&xb[base];
        const float4 x1 = *(const float4*)&xb[base + 4];
        float4 r0, r1;
        r0.x = x0.x + acc[nh * 4 + j][0] + boc;
        r0.y = x0.y + acc[nh * 4 + j][1] + boc;
        r0.z = x0.z + acc[nh * 4 + j][2] + boc;
        r0.w = x0.w + acc[nh * 4 + j][3] + boc;
        r1.x = x1.x + acc[nh * 4 + j][4] + boc;
        r1.y = x1.y + acc[nh * 4 + j][5] + boc;
        r1.z = x1.z + acc[nh * 4 + j][6] + boc;
        r1.w = x1.w + acc[nh * 4 + j][7] + boc;
        *(float4*)&res[(size_t)b * Cc * Tc + base]     = r0;
        *(float4*)&res[(size_t)b * Cc * Tc + base + 4] = r1;
      }
  }
}

// ---------------------------------------------------------------------------
// k2: depthwise conv1d(k=7,pad=3) + AdaLayerNorm -> y (B*T x C) bf16
// One wave per 32 t-values; each lane owns one t (LN is per-thread over C).
// ---------------------------------------------------------------------------
__global__ __launch_bounds__(32) void k2_convln(const float* __restrict__ res,
                                                const float* __restrict__ dww,
                                                const float* __restrict__ dwb,
                                                const int* __restrict__ cond,
                                                const float* __restrict__ sce,
                                                const float* __restrict__ she,
                                                bf16* __restrict__ yout) {
  __shared__ bf16 sC[Cc * 33];  // [c][t], stride 33 (66 B) -> conflict-free transpose
  const int b = blockIdx.y;
  const int t0 = blockIdx.x * 32;
  const int tid = threadIdx.x;
  const int t = t0 + tid;
  const float* rb = res + (size_t)b * Cc * Tc;

  float sum = 0.f, ssq = 0.f;
  for (int c = 0; c < Cc; ++c) {
    float a = dwb[c];
#pragma unroll
    for (int k = 0; k < 7; ++k) {
      const int tt = t + k - 3;
      const float xv = (tt >= 0 && tt < Tc) ? rb[(size_t)c * Tc + tt] : 0.f;
      a = fmaf(xv, dww[c * 7 + k], a);
    }
    sum += a;
    ssq = fmaf(a, a, ssq);
    sC[c * 33 + tid] = (bf16)a;
  }
  const float mu   = sum * (1.f / Cc);
  const float var  = ssq * (1.f / Cc) - mu * mu;
  const float rstd = rsqrtf(var + 1e-6f);
  const int e = cond[b];
  const float* sc = sce + e * Cc;
  const float* sh = she + e * Cc;
  for (int c = 0; c < Cc; ++c) {
    const float a = (float)sC[c * 33 + tid];
    sC[c * 33 + tid] = (bf16)(fmaf((a - mu) * rstd, sc[c], sh[c]));
  }
  __syncthreads();
  // transposed, coalesced 4B stores of normalized rows
  bf16* yo = yout + ((size_t)b * Tc + t0) * Cc;
  for (int r = 0; r < 32; ++r)
    for (int cc = 0; cc < 8; ++cc) {
      const int c = cc * 64 + tid * 2;
      v2bf pk;
      pk[0] = sC[c * 33 + r];
      pk[1] = sC[(c + 1) * 33 + r];
      *(v2bf*)&yo[(size_t)r * Cc + c] = pk;
    }
}

// ---------------------------------------------------------------------------
// k3: hid = GELU(y @ W1 + b1)   (65536x512)@(512x1536), bf16 in / f32 acc
// Block tile 64x256; 8 waves x 8 accumulators.
// ---------------------------------------------------------------------------
__global__ __launch_bounds__(256) void k3_mlp1(const bf16* __restrict__ y,
                                               const bf16* __restrict__ W1P,
                                               const float* __restrict__ b1,
                                               bf16* __restrict__ hid) {
  __shared__ alignas(32) bf16 sA[4 * 512];   //  4 KB: 4 A-fragments (64m x 32k)
  __shared__ alignas(32) bf16 sB[16 * 512];  // 16 KB: 16 B-fragments (32k x 256n)
  const int row0 = blockIdx.x * 64;
  const int n0   = blockIdx.y * 256;
  const int tid = threadIdx.x, lane = tid & 31, wave = tid >> 5;
  const int fm = wave & 3;
  const int ng = (wave >> 2) * 8;

  v8f acc[8] = {};
  for (int kk = 0; kk < 16; ++kk) {
    __syncthreads();
    {  // 16 B chunks: 8 contiguous k -> 8 contiguous fragment elements
      const int m = tid >> 2, k8 = (tid & 3) * 8;
      *(v8bf*)&sA[(m >> 4) * 512 + frag_slot(k8, m & 15)] =
          *(const v8bf*)&y[(size_t)(row0 + m) * Cc + kk * 32 + k8];
    }
    {
      const bf16* src = W1P + (((size_t)kk * (Ic >> 4) + (n0 >> 4)) << 9);  // 16 KB panel
      for (int i = tid; i < 512; i += 256)
        *(v16bf*)(sB + i * 16) = *(const v16bf*)(src + i * 16);
    }
    if (kk + 1 < 16)
      __builtin_prefetch(W1P + (((size_t)(kk + 1) * (Ic >> 4) + (n0 >> 4)) << 9) + tid * 32,
                         0, 1);
    __syncthreads();
    v16bf af = frag_ld(sA, fm, lane);
#pragma unroll
    for (int j = 0; j < 8; ++j) {
      v16bf bfr = frag_ld(sB, ng + j, lane);
      acc[j] = WMMA_BF16(af, bfr, acc[j]);
    }
  }
  const int mb = fm * 16 + 8 * (lane >> 4);
#pragma unroll
  for (int j = 0; j < 8; ++j) {
    const int n = n0 + (ng + j) * 16 + (lane & 15);
    const float bn = b1[n];
#pragma unroll
    for (int r = 0; r < 8; ++r) {
      const float v = acc[j][r] + bn;
      const float g = 0.5f * v * (1.f + erff(v * 0.70710678118f));  // exact GELU
      hid[(size_t)(row0 + mb + r) * Ic + n] = (bf16)g;
    }
  }
}

// ---------------------------------------------------------------------------
// k4: d_out(B,C,T) += transpose(hid @ W2 + b2)   (65536x1536)@(1536x512)
// ---------------------------------------------------------------------------
__global__ __launch_bounds__(256) void k4_mlp2(const bf16* __restrict__ hid,
                                               const bf16* __restrict__ W2P,
                                               const float* __restrict__ b2,
                                               float* __restrict__ out) {
  __shared__ alignas(32) bf16 sA[4 * 512];
  __shared__ alignas(32) bf16 sB[16 * 512];
  const int row0 = blockIdx.x * 64;
  const int n0   = blockIdx.y * 256;
  const int tid = threadIdx.x, lane = tid & 31, wave = tid >> 5;
  const int fm = wave & 3;
  const int ng = (wave >> 2) * 8;

  v8f acc[8] = {};
  for (int kk = 0; kk < Ic / 32; ++kk) {  // 48 K-chunks
    __syncthreads();
    {
      const int m = tid >> 2, k8 = (tid & 3) * 8;
      *(v8bf*)&sA[(m >> 4) * 512 + frag_slot(k8, m & 15)] =
          *(const v8bf*)&hid[(size_t)(row0 + m) * Ic + kk * 32 + k8];
    }
    {
      const bf16* src = W2P + (((size_t)kk * (Cc >> 4) + (n0 >> 4)) << 9);
      for (int i = tid; i < 512; i += 256)
        *(v16bf*)(sB + i * 16) = *(const v16bf*)(src + i * 16);
    }
    if (kk + 1 < Ic / 32)
      __builtin_prefetch(W2P + (((size_t)(kk + 1) * (Cc >> 4) + (n0 >> 4)) << 9) + tid * 32,
                         0, 1);
    __syncthreads();
    v16bf af = frag_ld(sA, fm, lane);
#pragma unroll
    for (int j = 0; j < 8; ++j) {
      v16bf bfr = frag_ld(sB, ng + j, lane);
      acc[j] = WMMA_BF16(af, bfr, acc[j]);
    }
  }
  const int mb = fm * 16 + 8 * (lane >> 4);
#pragma unroll
  for (int j = 0; j < 8; ++j) {
    const int c = n0 + (ng + j) * 16 + (lane & 15);
    const float bc = b2[c];
    const int rg0 = row0 + mb;              // 8 consecutive rows, same batch
    const int bb = rg0 >> 12;               // T = 4096
    const int t  = rg0 & (Tc - 1);
    float* po = out + ((size_t)bb * Cc + c) * Tc + t;  // 32 B aligned
    float4 o0 = *(const float4*)po;
    float4 o1 = *(const float4*)(po + 4);
    o0.x += acc[j][0] + bc; o0.y += acc[j][1] + bc;
    o0.z += acc[j][2] + bc; o0.w += acc[j][3] + bc;
    o1.x += acc[j][4] + bc; o1.y += acc[j][5] + bc;
    o1.z += acc[j][6] + bc; o1.w += acc[j][7] + bc;
    *(float4*)po       = o0;  // residual add
    *(float4*)(po + 4) = o1;
  }
}

// ---------------------------------------------------------------------------
extern "C" void kernel_launch(void* const* d_in, const int* in_sizes, int n_in,
                              void* d_out, int out_size, void* d_ws, size_t ws_size,
                              hipStream_t stream) {
  const float* x   = (const float*)d_in[0];
  const int*   cid = (const int*)  d_in[1];
  const float* Wq  = (const float*)d_in[2];
  const float* bq  = (const float*)d_in[3];
  const float* Wkv = (const float*)d_in[4];
  const float* bkv = (const float*)d_in[5];
  const float* Wo  = (const float*)d_in[6];
  const float* bo  = (const float*)d_in[7];
  const float* dww = (const float*)d_in[8];
  const float* dwb = (const float*)d_in[9];
  const float* sce = (const float*)d_in[10];
  const float* she = (const float*)d_in[11];
  const float* W1  = (const float*)d_in[12];
  const float* b1  = (const float*)d_in[13];
  const float* W2  = (const float*)d_in[14];
  const float* b2  = (const float*)d_in[15];
  const float* aux = (const float*)d_in[16];
  float* out = (float*)d_out;

  char* ws = (char*)d_ws;
  bf16* kr  = (bf16*)(ws);                    //  64 KB
  bf16* vm  = (bf16*)(ws + 65536);            //  64 KB
  bf16* WqP = (bf16*)(ws + 131072);           // 512 KB
  bf16* WoP = (bf16*)(ws + 655360);           // 512 KB
  bf16* W1P = (bf16*)(ws + 1179648);          // 1.5 MB
  bf16* W2P = (bf16*)(ws + 2752512);          // 1.5 MB
  bf16* yn  = (bf16*)(ws + 4325376);          //  64 MB
  bf16* hid = (bf16*)(ws + 4325376 + (size_t)Bc * Tc * Cc * 2);  // 192 MB

  // one-time (per launch) weight pre-swizzle into WMMA fragment panels
  kp_w<<<dim3((Cc * Cc) / 256), dim3(256), 0, stream>>>(Wq, WqP, Cc, Cc);
  kp_w<<<dim3((Cc * Cc) / 256), dim3(256), 0, stream>>>(Wo, WoP, Cc, Cc);
  kp_w<<<dim3((Cc * Ic) / 256), dim3(256), 0, stream>>>(W1, W1P, Cc, Ic);
  kp_w<<<dim3((Ic * Cc) / 256), dim3(256), 0, stream>>>(W2, W2P, Ic, Cc);

  k0_kv  <<<dim3((Lc * 2 * Cc) / 256), dim3(256), 0, stream>>>(aux, Wkv, bkv, kr, vm);
  k1_attn<<<dim3(Tc / 32, Bc),         dim3(256), 0, stream>>>(x, WqP, bq, WoP, bo, kr, vm, out);
  k2_convln<<<dim3(Tc / 32, Bc),       dim3(32),  0, stream>>>(out, dww, dwb, cid, sce, she, yn);
  k3_mlp1<<<dim3((Bc * Tc) / 64, Ic / 256), dim3(256), 0, stream>>>(yn, W1P, b1, hid);
  k4_mlp2<<<dim3((Bc * Tc) / 64, Cc / 256), dim3(256), 0, stream>>>(hid, W2P, b2, out);
}
